// BitLinear_49340584296865
// MI455X (gfx1250) — matrix operand order repaired and checked by reading, not compiled, not run on hardware
//
#include <hip/hip_runtime.h>

typedef __attribute__((ext_vector_type(16))) _Float16 v16h;
typedef __attribute__((ext_vector_type(8)))  _Float16 v8h;
typedef __attribute__((ext_vector_type(4)))  _Float16 v4h;
typedef __attribute__((ext_vector_type(8)))  float    v8f;
typedef __attribute__((ext_vector_type(4)))  float    v4f;
typedef __attribute__((ext_vector_type(2)))  int      v2i;

#define IN_F   4096
#define OUT_F  11008
#define M_ROWS 64
#define KCHUNK 256
#define LDS_STRIDE 264                      // halves; 528B row stride => bank rotation 4/row
#define BUF_HALVES (M_ROWS * LDS_STRIDE)    // one staging buffer (33KB)

// ---- async copy global -> LDS (no VGPR data), ASYNCcnt-tracked ----
// gfx12 generic pointers to LDS carry the LDS byte offset in the low 32 bits.
__device__ __forceinline__ void async_copy_b128(_Float16* lds_dst, const _Float16* gsrc) {
  const unsigned ldsoff = (unsigned)(unsigned long long)(const void*)lds_dst;
  asm volatile("global_load_async_to_lds_b128 %0, %1, off"
               :: "v"(ldsoff), "v"(gsrc));
}
__device__ __forceinline__ void wait_async_all() {
  asm volatile("s_wait_asynccnt 0x0" ::: "memory");
}

// ---- kernel 1: one-shot x fp32 -> fp16 (into d_ws); also warms L2 with xh ----
__global__ __launch_bounds__(256) void x_to_f16(const float* __restrict__ x,
                                                _Float16* __restrict__ xh) {
  const int idx = (blockIdx.x * 256 + threadIdx.x) * 4;   // 65536 threads * 4 floats
  v4f v = *(const v4f*)(x + idx);
  v4h h;
  h.x = (_Float16)v.x; h.y = (_Float16)v.y;
  h.z = (_Float16)v.z; h.w = (_Float16)v.w;
  *(v4h*)(xh + idx) = h;
}

// ---- kernel 2: BitLinear GEMM; wave owns one 16-col N-tile and all four 16-row M-tiles ----
template <int FUSED>
__global__ __launch_bounds__(128) void bitlinear_wmma_f16(
    const void* __restrict__ xsrc,   // FUSED: float*  else: _Float16*
    const int*  __restrict__ bp,
    float*      __restrict__ out)
{
  __shared__ _Float16 lds_x[2 * BUF_HALVES];   // double-buffered x chunk

  const int t    = threadIdx.x;
  const int lane = t & 31;
  const int wave = t >> 5;          // 0..3 -> N-tile within block
  const int l15  = lane & 15;
  const int lhi  = lane >> 4;       // 0 or 1

  const int  ncol      = blockIdx.x * 64 + wave * 16 + l15;  // output column owned by this lane
  const long wrow_base = (long)ncol * (IN_F / 8);            // int32 index of W-row's packed bytes

  v8f acc[4] = {};                  // 4 M-tiles of 16x16

  const _Float16* xh = (const _Float16*)xsrc;
  const float*    xf = (const float*)xsrc;

  // ---- prologue: async-stage chunk 0 into buffer 0 ----
  if (!FUSED) {
#pragma unroll
    for (int i = 0; i < 16; ++i) {
      const int idx = i * 128 + t;               // 2048 b128 slots
      const int row = idx >> 5, c8 = idx & 31;
      async_copy_b128(lds_x + row * LDS_STRIDE + c8 * 8,
                      xh + row * IN_F + c8 * 8);
    }
    wait_async_all();
    __syncthreads();
  }

  int buf = 0;
  for (int k0 = 0; k0 < IN_F; k0 += KCHUNK) {
    const bool more = (k0 + KCHUNK < IN_F);

    if (FUSED) {
      // simple fallback staging (fp32 -> fp16 in-kernel), single buffer
      __syncthreads();
#pragma unroll
      for (int i = 0; i < 32; ++i) {
        const int idx = i * 128 + t;             // 4096 float4 slots
        const int row = idx >> 6, c4 = idx & 63;
        v4f v = *(const v4f*)(xf + row * IN_F + k0 + c4 * 4);
        v4h h;
        h.x = (_Float16)v.x; h.y = (_Float16)v.y;
        h.z = (_Float16)v.z; h.w = (_Float16)v.w;
        *(v4h*)(lds_x + row * LDS_STRIDE + c4 * 4) = h;
      }
      __syncthreads();
    } else if (more) {
      // issue async copies of the NEXT chunk into the other buffer; wait after compute
      _Float16* sbase = lds_x + (buf ^ 1) * BUF_HALVES;
#pragma unroll
      for (int i = 0; i < 16; ++i) {
        const int idx = i * 128 + t;
        const int row = idx >> 5, c8 = idx & 31;
        async_copy_b128(sbase + row * LDS_STRIDE + c8 * 8,
                        xh + row * IN_F + (k0 + KCHUNK) + c8 * 8);
      }
    }

    // prefetch next K-chunk of packed weights (one 128B line covers the row's whole chunk)
    if (more)
      __builtin_prefetch(bp + wrow_base + ((k0 + KCHUNK) >> 3), 0, 0);

    const _Float16* lbase     = lds_x + buf * BUF_HALVES;
    const _Float16* arow_base = lbase + l15 * LDS_STRIDE + lhi * 16;

    // ---- hoist this chunk's 8 weight loads: issue together, drain under compute ----
    v2i wbs[8];
#pragma unroll
    for (int s = 0; s < 8; ++s)
      wbs[s] = *(const v2i*)(bp + wrow_base + ((k0 + s * 32 + lhi * 16) >> 3));

#pragma unroll
    for (int s = 0; s < 8; ++s) {
      const int ks = s * 32;

      // ---- B fragment (32x16 f16): unpack 16 K-bits from pre-loaded registers ----
      const unsigned nv = ~(((unsigned)(wbs[s].x & 0xFF) << 8) | (unsigned)(wbs[s].y & 0xFF));
      union { v16h v; unsigned p[8]; } B;
#pragma unroll
      for (int j = 0; j < 8; ++j) {
        const unsigned w = nv << (2 * j);   // bit15 = inv(kk=2j), bit14 = inv(kk=2j+1)
        B.p[j] = 0x3C003C00u | (w & 0x8000u) | ((w << 17) & 0x80000000u);
      }

      // ---- load ALL four A fragments first (distinct live ranges => pipelined ds loads) ----
      union { v16h v; v8h h[2]; } A0, A1, A2, A3;
      A0.h[0] = *(const v8h*)(arow_base + 0 * 16 * LDS_STRIDE + ks);
      A0.h[1] = *(const v8h*)(arow_base + 0 * 16 * LDS_STRIDE + ks + 8);
      A1.h[0] = *(const v8h*)(arow_base + 1 * 16 * LDS_STRIDE + ks);
      A1.h[1] = *(const v8h*)(arow_base + 1 * 16 * LDS_STRIDE + ks + 8);
      A2.h[0] = *(const v8h*)(arow_base + 2 * 16 * LDS_STRIDE + ks);
      A2.h[1] = *(const v8h*)(arow_base + 2 * 16 * LDS_STRIDE + ks + 8);
      A3.h[0] = *(const v8h*)(arow_base + 3 * 16 * LDS_STRIDE + ks);
      A3.h[1] = *(const v8h*)(arow_base + 3 * 16 * LDS_STRIDE + ks + 8);

      // ---- 4 back-to-back WMMAs share one B ----
      acc[0] = __builtin_amdgcn_wmma_f32_16x16x32_f16(false, A0.v, false, B.v, (short)0, acc[0], false, false);
      acc[1] = __builtin_amdgcn_wmma_f32_16x16x32_f16(false, A1.v, false, B.v, (short)0, acc[1], false, false);
      acc[2] = __builtin_amdgcn_wmma_f32_16x16x32_f16(false, A2.v, false, B.v, (short)0, acc[2], false, false);
      acc[3] = __builtin_amdgcn_wmma_f32_16x16x32_f16(false, A3.v, false, B.v, (short)0, acc[3], false, false);
    }

    // ---- commit next chunk: drain async copies, swap buffers ----
    if (!FUSED && more) {
      wait_async_all();
      __syncthreads();
      buf ^= 1;
    }
  }

  // ---- store C: VGPR r, lane<16 -> M=r, lane>=16 -> M=8+r; N = lane&15 ----
#pragma unroll
  for (int mt = 0; mt < 4; ++mt)
#pragma unroll
    for (int r = 0; r < 8; ++r)
      out[(long)(mt * 16 + lhi * 8 + r) * OUT_F + ncol] = acc[mt][r];
}

extern "C" void kernel_launch(void* const* d_in, const int* in_sizes, int n_in,
                              void* d_out, int out_size, void* d_ws, size_t ws_size,
                              hipStream_t stream) {
  (void)in_sizes; (void)n_in; (void)out_size;
  const float* x   = (const float*)d_in[0];
  const int*   bp  = (const int*)d_in[1];
  float*       out = (float*)d_out;

  const size_t need = (size_t)M_ROWS * IN_F * sizeof(_Float16);  // 512 KB
  dim3 grid(OUT_F / 64);   // 172 blocks
  dim3 block(128);         // 4 waves: one 16-col N-tile each, four 16-row M-tiles

  if (ws_size >= need) {
    _Float16* xh = (_Float16*)d_ws;
    x_to_f16<<<dim3(256), dim3(256), 0, stream>>>(x, xh);        // 64*4096 elements
    bitlinear_wmma_f16<0><<<grid, block, 0, stream>>>((const void*)xh, bp, out);
  } else {
    bitlinear_wmma_f16<1><<<grid, block, 0, stream>>>((const void*)x, bp, out);
  }
}
